// ClusterBasedContrastiveLoss_64484638982544
// MI455X (gfx1250) — matrix operand order repaired and checked by reading, not compiled, not run on hardware
//
#include <hip/hip_runtime.h>
#include <hip/hip_bf16.h>
#include <math.h>

typedef __attribute__((ext_vector_type(16))) _Float16 v16h;
typedef __attribute__((ext_vector_type(8)))  float    v8f;

#define B_N    16384
#define C_N    50
#define K_N    128
#define D_N    128
#define ROWS   (C_N * 2 * K_N)   /* 12800 */
#define TEMP_INV 2.0f            /* 1/TEMP */
#define EPS_F  1e-8f

// ---------------------------------------------------------------------------
// Phase 1: exact top-K per cluster via 4-pass radix select on float bits.
// prob is uniform[0,1) -> non-negative -> uint bit pattern is order-preserving.
// One workgroup (256 threads) per cluster; histogram in LDS.
// ---------------------------------------------------------------------------
__global__ void topk_kernel(const float* __restrict__ prob, int* __restrict__ topk) {
  const int c = blockIdx.x;
  const int tid = threadIdx.x;
  __shared__ unsigned hist[256];
  __shared__ unsigned s_prefix, s_need, s_cnt;
  if (tid == 0) { s_prefix = 0u; s_need = K_N; s_cnt = 0u; }
  __syncthreads();
  for (int pass = 0; pass < 4; ++pass) {
    const int shift = 24 - pass * 8;
    hist[tid] = 0u;
    __syncthreads();
    const unsigned prefix = s_prefix;
    for (int b = tid; b < B_N; b += 256) {
      const unsigned key = __float_as_uint(prob[(size_t)b * C_N + c]);
      const unsigned hi = (pass == 0) ? 0u : (key >> (shift + 8));
      if (hi == prefix) atomicAdd(&hist[(key >> shift) & 0xFFu], 1u);
    }
    __syncthreads();
    if (tid == 0) {
      unsigned need = s_need, cum = 0u; int bin = 0;
      for (int bb = 255; bb >= 0; --bb) {
        if (cum + hist[bb] >= need) { bin = bb; break; }
        cum += hist[bb];
      }
      s_prefix = (prefix << 8) | (unsigned)bin;
      s_need = need - cum;
    }
    __syncthreads();
  }
  const unsigned T = s_prefix;  // exact bit pattern of the K-th largest value
  for (int b = tid; b < B_N; b += 256) {
    const unsigned key = __float_as_uint(prob[(size_t)b * C_N + c]);
    if (key > T) { const unsigned pos = atomicAdd(&s_cnt, 1u); topk[c * K_N + pos] = b; }
  }
  __syncthreads();  // all strictly-greater entries placed first
  for (int b = tid; b < B_N; b += 256) {
    const unsigned key = __float_as_uint(prob[(size_t)b * C_N + c]);
    if (key == T) {
      const unsigned pos = atomicAdd(&s_cnt, 1u);
      if (pos < K_N) topk[c * K_N + pos] = b;
    }
  }
}

// ---------------------------------------------------------------------------
// Phase 2: gather + L2-normalize selected rows; store as f16 matrix P (12800x128).
// One wave per row (wave32), float4 loads, full-wave shfl_xor reduction.
// ---------------------------------------------------------------------------
__global__ void gather_norm_kernel(const float* __restrict__ z_i,
                                   const float* __restrict__ z_j,
                                   const int* __restrict__ topk,
                                   _Float16* __restrict__ P) {
  const int lane = threadIdx.x & 31;
  const int wid  = threadIdx.x >> 5;
  const int row  = blockIdx.x * 8 + wid;
  const int c = row >> 8;      // 256 rows per cluster
  const int p = row & 255;
  const float* src;
  if (p < K_N) src = z_i + (size_t)topk[c * K_N + p] * D_N;
  else         src = z_j + (size_t)topk[c * K_N + (p - K_N)] * D_N;
  const float4 v = reinterpret_cast<const float4*>(src)[lane];
  float ss = v.x * v.x + v.y * v.y + v.z * v.z + v.w * v.w;
  for (int off = 16; off > 0; off >>= 1) ss += __shfl_xor(ss, off);
  const float inv = 1.0f / fmaxf(sqrtf(ss), EPS_F);
  _Float16* dst = P + (size_t)row * D_N + lane * 4;
  dst[0] = (_Float16)(v.x * inv);
  dst[1] = (_Float16)(v.y * inv);
  dst[2] = (_Float16)(v.z * inv);
  dst[3] = (_Float16)(v.w * inv);
}

__global__ void zero_kernel(float* __restrict__ p, int n) {
  const int i = blockIdx.x * blockDim.x + threadIdx.x;
  if (i < n) p[i] = 0.0f;
}

// ---------------------------------------------------------------------------
// Phase 3: fused S = exp(2 * P P^T) with masked row-sums, WMMA f16->f32.
// Grid 100x100; each WG computes a 128x128 tile of S (never materialized).
// Wave w owns rows [w*16, w*16+16) x 8 col-tiles; K streamed in 4 slices of 32
// (= one v_wmma_f32_16x16x32_f16 per tile per slice) through padded LDS.
// ---------------------------------------------------------------------------
__global__ void __launch_bounds__(256)
sim_gemm_kernel(const _Float16* __restrict__ P,
                float* __restrict__ sumAll,
                float* __restrict__ ownSum,
                float* __restrict__ posSum) {
  constexpr int LDP = 40;  // row stride in halves: 80B, 16B-aligned, conflict-padded
  __shared__ alignas(16) _Float16 Als[128][LDP];
  __shared__ alignas(16) _Float16 Bls[128][LDP];
  const int rb = blockIdx.x, cb = blockIdx.y;
  const int rowbase = rb * 128, colbase = cb * 128;
  const bool isOwn = (rb >> 1) == (cb >> 1);          // same cluster block
  const bool isPos = isOwn && ((cb & 1) == 0);        // first K cols of cluster
  const int tid  = threadIdx.x;
  const int lane = tid & 31;
  const int w    = tid >> 5;       // wave id 0..7
  const int hi   = lane >> 4;      // lane half
  const int lm   = lane & 15;

  v8f acc[8] = {};                 // 8 col-tiles of f32 accumulators

  for (int kc = 0; kc < 4; ++kc) {
    __syncthreads();
    // Stage both 128x32 f16 panels (A rows, B cols-as-rows) via 16B chunks.
    for (int i = tid; i < 512; i += 256) {
      const int r = i >> 2, j = i & 3;
      const size_t ga = (size_t)(rowbase + r) * D_N + kc * 32 + j * 8;
      const size_t gb = (size_t)(colbase + r) * D_N + kc * 32 + j * 8;
      *reinterpret_cast<uint4*>(&Als[r][j * 8]) = *reinterpret_cast<const uint4*>(P + ga);
      *reinterpret_cast<uint4*>(&Bls[r][j * 8]) = *reinterpret_cast<const uint4*>(P + gb);
    }
    __syncthreads();

    // A fragment (16x32 f16): lane holds row M=lm of this wave's stripe.
    v16h a;
    const int arow = w * 16 + lm;
    #pragma unroll
    for (int v = 0; v < 8; ++v) {
      const int kb = (v < 4) ? (2 * v + 8 * hi) : (16 + 2 * (v - 4) + 8 * hi);
      a[2 * v]     = Als[arow][kb];
      a[2 * v + 1] = Als[arow][kb + 1];
    }
    #pragma unroll
    for (int tn = 0; tn < 8; ++tn) {
      // B fragment (32x16 f16): B[k][n] = P[colbase + tn*16 + n][k].
      v16h b;
      const int brow = tn * 16 + lm;  // N = lm
      #pragma unroll
      for (int v = 0; v < 8; ++v) {
        const int kb = (v < 4) ? (2 * v + 8 * hi) : (16 + 2 * (v - 4) + 8 * hi);
        b[2 * v]     = Bls[brow][kb];
        b[2 * v + 1] = Bls[brow][kb + 1];
      }
      acc[tn] = __builtin_amdgcn_wmma_f32_16x16x32_f16(
          /*neg_a=*/false, a, /*neg_b=*/false, b,
          /*c_mod=*/(short)0, acc[tn], /*reuse_a=*/false, /*reuse_b=*/false);
    }
  }

  // Epilogue: e = exp(dot/TEMP); accumulate per-lane partial row sums over N.
  float sA[8], sO[8], sP[8];
  #pragma unroll
  for (int i = 0; i < 8; ++i) { sA[i] = 0.0f; sO[i] = 0.0f; sP[i] = 0.0f; }
  #pragma unroll
  for (int tn = 0; tn < 8; ++tn) {
    #pragma unroll
    for (int i = 0; i < 8; ++i) {
      const float e = __expf(acc[tn][i] * TEMP_INV);
      sA[i] += e;
      if (isOwn) sO[i] += e;
      if (isPos) sP[i] += e;
    }
  }
  // Reduce over the 16 N-lanes (masks < 16 stay within each half-wave group).
  #pragma unroll
  for (int i = 0; i < 8; ++i) {
    #pragma unroll
    for (int off = 1; off < 16; off <<= 1) {
      sA[i] += __shfl_xor(sA[i], off);
      sO[i] += __shfl_xor(sO[i], off);
      sP[i] += __shfl_xor(sP[i], off);
    }
  }
  if (lm == 0) {
    #pragma unroll
    for (int i = 0; i < 8; ++i) {
      const int row = rowbase + w * 16 + 8 * hi + i;  // C/D layout: M = i + 8*hi
      atomicAdd(&sumAll[row], sA[i]);
      if (isOwn) atomicAdd(&ownSum[row], sO[i]);
      if (isPos) posSum[row] = sP[i];   // exactly one WG per row writes pos
    }
  }
}

// ---------------------------------------------------------------------------
// Phase 4: loss = mean(log(sum_all - own) - log(pos))
// ---------------------------------------------------------------------------
__global__ void finalize_kernel(const float* __restrict__ sumAll,
                                const float* __restrict__ ownSum,
                                const float* __restrict__ posSum,
                                float* __restrict__ out) {
  __shared__ float red[256];
  const int tid = threadIdx.x;
  float acc = 0.0f;
  for (int r = tid; r < ROWS; r += 256)
    acc += logf(sumAll[r] - ownSum[r]) - logf(posSum[r]);
  red[tid] = acc;
  __syncthreads();
  for (int s = 128; s > 0; s >>= 1) {
    if (tid < s) red[tid] += red[tid + s];
    __syncthreads();
  }
  if (tid == 0) out[0] = red[0] / (float)ROWS;
}

extern "C" void kernel_launch(void* const* d_in, const int* in_sizes, int n_in,
                              void* d_out, int out_size, void* d_ws, size_t ws_size,
                              hipStream_t stream) {
  (void)in_sizes; (void)n_in; (void)out_size; (void)ws_size;
  const float* prob = (const float*)d_in[0];
  const float* z_i  = (const float*)d_in[1];
  const float* z_j  = (const float*)d_in[2];
  float* out = (float*)d_out;

  // Workspace layout (all 16B-aligned): 3 row-sum arrays, topk indices, P(f16).
  char* ws = (char*)d_ws;
  float* sumAll = (float*)ws;                    // ROWS floats
  float* ownSum = sumAll + ROWS;                 // ROWS floats
  float* posSum = ownSum + ROWS;                 // ROWS floats
  int*   topk   = (int*)(ws + (size_t)3 * ROWS * sizeof(float));          // C*K ints
  _Float16* P   = (_Float16*)(ws + (size_t)3 * ROWS * sizeof(float)
                                  + (size_t)C_N * K_N * sizeof(int));     // ROWS*D halves

  topk_kernel<<<C_N, 256, 0, stream>>>(prob, topk);
  zero_kernel<<<(3 * ROWS + 255) / 256, 256, 0, stream>>>(sumAll, 3 * ROWS);
  gather_norm_kernel<<<ROWS / 8, 256, 0, stream>>>(z_i, z_j, topk, P);
  sim_gemm_kernel<<<dim3(ROWS / 128, ROWS / 128), 256, 0, stream>>>(P, sumAll, ownSum, posSum);
  finalize_kernel<<<1, 256, 0, stream>>>(sumAll, ownSum, posSum, out);
}
